// DepthNet_386547057036
// MI455X (gfx1250) — matrix operand order repaired and checked by reading, not compiled, not run on hardware
//
#include <hip/hip_runtime.h>
#include <hip/hip_bf16.h>

typedef __attribute__((ext_vector_type(16))) _Float16 v16h;
typedef __attribute__((ext_vector_type(8)))  _Float16 v8h;
typedef __attribute__((ext_vector_type(8)))  float    v8f;

#define HH 384
#define WW 384
#define PW 388              // padded width (pad=2 each side for 5x5 SAME)
#define PP (PW*PW)          // 150544 padded plane pixels

// ---------------- workspace layout (bytes, all 256-aligned) ----------------
// xpc : (PP+PW)*16 f16 channel-last input (+1 zero row for tap-25 pad)  4,829,824 B
// hp1c: PP*32 f16 channel-last conv1 output                             9,634,816 B
// hp2c: PP*8  f16 channel-last conv2 output                             2,408,704 B
// wb1 : 13*2*32*16 f16 conv1 weights, WMMA-B fragment order                26,624 B
// wb2 : 25*32*16  f16 conv2 weights, WMMA-B fragment order                 25,600 B
// par : folded BN params                                                    4,096 B
// dep : 147456 u8 depth bins
#define O_XPC  0
#define O_HP1  4829952
#define O_HP2  14464768
#define O_WB1  16873472
#define O_WB2  16900096
#define O_PAR  16925696
#define O_DEP  16929792

// par layout: [0..31]=bias1' [32..39]=bias2' [40]=bias3' [41..240]=w3*scale3
//             [256..287]=scale1 [288..295]=scale2 [296]=scale3

#define WMMA_F16(A, B, C) \
  __builtin_amdgcn_wmma_f32_16x16x32_f16(false, (A), false, (B), (short)0, (C), false, false)

__global__ void dn_prep_params(const float* c1_b, const float* g1, const float* b1,
                               const float* m1, const float* v1,
                               const float* c2_b, const float* g2, const float* b2,
                               const float* m2, const float* v2,
                               const float* c3_w, const float* c3_b, const float* g3,
                               const float* b3, const float* m3, const float* v3,
                               float* par) {
  int t = threadIdx.x;
  if (t < 32) {
    float s = g1[t] * rsqrtf(v1[t] + 1e-5f);
    par[256 + t] = s;
    par[t] = (c1_b[t] - m1[t]) * s + b1[t];
  }
  if (t < 8) {
    float s = g2[t] * rsqrtf(v2[t] + 1e-5f);
    par[288 + t] = s;
    par[32 + t] = (c2_b[t] - m2[t]) * s + b2[t];
  }
  float s3 = g3[0] * rsqrtf(v3[0] + 1e-5f);
  if (t == 0) {
    par[296] = s3;
    par[40] = (c3_b[0] - m3[0]) * s3 + b3[0];
  }
  for (int k = t; k < 200; k += blockDim.x) par[41 + k] = c3_w[k] * s3;
}

// channel-last zero-padded f16 input: xpc[pixel][16], channels 0..8 = concat(l,a,n)
__global__ void dn_pack_xpc(const float* __restrict__ light, const float* __restrict__ albedo,
                            const float* __restrict__ normals, _Float16* __restrict__ xpc) {
  int i = blockIdx.x * blockDim.x + threadIdx.x;
  if (i >= (PP + PW) * 16) return;
  int pix = i >> 4, c = i & 15;
  int a = pix / PW, b = pix - a * PW;           // a may be 388 (extra pad row)
  float v = 0.f;
  if (c < 9 && a >= 2 && a < 2 + HH && b >= 2 && b < 2 + WW) {
    int p = (a - 2) * WW + (b - 2);
    const float* src = (c < 3) ? light : (c < 6 ? albedo : normals);
    int cc = (c < 3) ? c : (c < 6 ? c - 3 : c - 6);
    v = src[cc * HH * WW + p];
  }
  xpc[i] = (_Float16)v;
}

__global__ void dn_zero16(uint4* p, int n16) {
  int i = blockIdx.x * blockDim.x + threadIdx.x;
  if (i < n16) p[i] = uint4{0u, 0u, 0u, 0u};
}

// conv1 weights -> WMMA B-fragment order. GEMM K ordering: K = tap*16 + ci
// fragment element (kc,nt,L,j): K = kc*32 + (j>>3)*16 + (L>>4)*8 + (j&7) ; N = nt*16 + (L&15)
__global__ void dn_pack_wb1(const float* __restrict__ c1_w, const float* __restrict__ par,
                            _Float16* __restrict__ wb1) {
  int i = blockIdx.x * blockDim.x + threadIdx.x;
  if (i >= 13 * 2 * 32 * 16) return;
  int j = i & 15, L = (i >> 4) & 31, nt = (i >> 9) & 1, kc = i >> 10;
  int N = nt * 16 + (L & 15);
  int K = kc * 32 + ((j >> 3) << 4) + ((L >> 4) << 3) + (j & 7);
  int tap = K >> 4, ci = K & 15;
  float v = 0.f;
  if (tap < 25 && ci < 9) v = c1_w[N * 225 + ci * 25 + tap] * par[256 + N];
  wb1[i] = (_Float16)v;
}

// conv2 weights: K = tap*32 + ci (tap == kc exactly)
__global__ void dn_pack_wb2(const float* __restrict__ c2_w, const float* __restrict__ par,
                            _Float16* __restrict__ wb2) {
  int i = blockIdx.x * blockDim.x + threadIdx.x;
  if (i >= 25 * 32 * 16) return;
  int j = i & 15, L = (i >> 4) & 31, kc = i >> 9;
  int N = L & 15;
  int ci = ((j >> 3) << 4) + ((L >> 4) << 3) + (j & 7); // K within chunk
  float v = 0.f;
  if (N < 8) v = c2_w[N * 800 + ci * 25 + kc] * par[288 + N];
  wb2[i] = (_Float16)v;
}

// conv1: implicit GEMM, M=147456 px, N=32, K=26 taps * 16 ch = 13 chunks.
// A loads: two global_load_b128 per chunk (channel-last => 8 consecutive K in memory).
__global__ void __launch_bounds__(256) dn_conv1_wmma(const _Float16* __restrict__ xpc,
                                                     const _Float16* __restrict__ wb1,
                                                     const float* __restrict__ par,
                                                     _Float16* __restrict__ hp1c) {
  int wave = blockIdx.x * 8 + (threadIdx.x >> 5);
  int lane = threadIdx.x & 31;
  int p0 = wave << 4;                 // 16 px per wave, never crosses a row (384%16==0)
  int h = p0 / WW;
  int w0 = p0 - h * WW;
  int pix = h * PW + (w0 + (lane & 15));        // padded coords of lane's M row
  int khi = (lane >> 4) << 3;                   // lane-half K offset in A layout
  __builtin_prefetch(xpc + pix * 16, 0, 1);

  v8f acc0 = {}; v8f acc1 = {};
#pragma unroll
  for (int kc = 0; kc < 13; ++kc) {
    int t0 = 2 * kc, t1 = 2 * kc + 1;           // t1 == 25 (kc=12) reads zeroed pad row
    int ky0 = t0 / 5, kx0 = t0 % 5;
    int ky1 = t1 / 5, kx1 = t1 % 5;
    v8h alo = *(const v8h*)(xpc + (pix + ky0 * PW + kx0) * 16 + khi);
    v8h ahi = *(const v8h*)(xpc + (pix + ky1 * PW + kx1) * 16 + khi);
    v16h a = __builtin_shufflevector(alo, ahi, 0, 1, 2, 3, 4, 5, 6, 7,
                                     8, 9, 10, 11, 12, 13, 14, 15);
    v16h b0 = *(const v16h*)(wb1 + ((kc * 2 + 0) * 32 + lane) * 16);
    v16h b1 = *(const v16h*)(wb1 + ((kc * 2 + 1) * 32 + lane) * 16);
    acc0 = WMMA_F16(a, b0, acc0);
    acc1 = WMMA_F16(a, b1, acc1);
  }
  // C/D layout: lane L, VGPR r -> M = r + (L>>4)*8, N = L&15
  int ch = lane & 15;
  int mbase = (lane >> 4) << 3;
  float bA = par[ch];
  float bB = par[16 + ch];
#pragma unroll
  for (int r = 0; r < 8; ++r) {
    int dst = ((h + 2) * PW + (w0 + mbase + r + 2)) * 32;
    float y0 = acc0[r] + bA; y0 = y0 > 0.f ? y0 : 0.f;
    float y1 = acc1[r] + bB; y1 = y1 > 0.f ? y1 : 0.f;
    hp1c[dst + ch]      = (_Float16)y0;
    hp1c[dst + 16 + ch] = (_Float16)y1;
  }
}

// conv2: M=147456, N=16 (8 valid), K = 25 taps * 32 ch = 25 chunks, 2 b128 A-loads/chunk
__global__ void __launch_bounds__(256) dn_conv2_wmma(const _Float16* __restrict__ hp1c,
                                                     const _Float16* __restrict__ wb2,
                                                     const float* __restrict__ par,
                                                     _Float16* __restrict__ hp2c) {
  int wave = blockIdx.x * 8 + (threadIdx.x >> 5);
  int lane = threadIdx.x & 31;
  int p0 = wave << 4;
  int h = p0 / WW;
  int w0 = p0 - h * WW;
  int pix = h * PW + (w0 + (lane & 15));
  int khi = (lane >> 4) << 3;

  v8f acc = {};
#pragma unroll
  for (int kc = 0; kc < 25; ++kc) {
    int ky = kc / 5, kx = kc % 5;
    const _Float16* ap = hp1c + (pix + ky * PW + kx) * 32 + khi;
    v8h alo = *(const v8h*)(ap);                // channels khi..khi+7
    v8h ahi = *(const v8h*)(ap + 16);           // channels 16+khi..16+khi+7
    v16h a = __builtin_shufflevector(alo, ahi, 0, 1, 2, 3, 4, 5, 6, 7,
                                     8, 9, 10, 11, 12, 13, 14, 15);
    v16h b = *(const v16h*)(wb2 + (kc * 32 + lane) * 16);
    acc = WMMA_F16(a, b, acc);
  }
  int ch = lane & 15;
  int mbase = (lane >> 4) << 3;
  if (ch < 8) {
    float bs = par[32 + ch];
#pragma unroll
    for (int r = 0; r < 8; ++r) {
      int dst = ((h + 2) * PW + (w0 + mbase + r + 2)) * 8;
      float y = acc[r] + bs; y = y > 0.f ? y : 0.f;
      hp2c[dst + ch] = (_Float16)y;
    }
  }
}

// conv3 (8->1) + BN + sigmoid + depth binning; one b128 load per tap (channel-last)
__global__ void __launch_bounds__(256) dn_conv3_depth(const _Float16* __restrict__ hp2c,
                                                      const float* __restrict__ par,
                                                      unsigned char* __restrict__ depth) {
  __shared__ float w3t[200];                    // tap-major: w3t[tap*8+ci]
  __shared__ float b3s;
  int t = threadIdx.x;
  for (int k = t; k < 200; k += 256) {
    int tap = k >> 3, ci = k & 7;
    w3t[k] = par[41 + ci * 25 + tap];
  }
  if (t == 0) b3s = par[40];
  __syncthreads();
  int p = blockIdx.x * 256 + t;
  int h = p / WW, w = p - h * WW;
  float acc = b3s;
  int base = h * PW + w;
#pragma unroll
  for (int tap = 0; tap < 25; ++tap) {
    int ky = tap / 5, kx = tap % 5;
    v8h hv = *(const v8h*)(hp2c + (base + ky * PW + kx) * 8);
#pragma unroll
    for (int ci = 0; ci < 8; ++ci) acc += w3t[tap * 8 + ci] * (float)hv[ci];
  }
  float emb = 1.f / (1.f + expf(-acc));
  int d = (int)floorf(emb * (16.f - 1e-5f));
  d = d < 0 ? 0 : (d > 15 ? 15 : d);
  depth[p] = (unsigned char)d;
}

// Sparse reformulation of 3D conv + depth gather + final composition.
// 16x16 tile with 3-halo; depth/light halo + all f_w coeffs staged in LDS.
__global__ void __launch_bounds__(256) dn_final(const unsigned char* __restrict__ depth,
                                                const float* __restrict__ light,
                                                const float* __restrict__ albedo,
                                                const float* __restrict__ f_w,
                                                const float* __restrict__ f_b,
                                                float* __restrict__ out) {
  __shared__ float fw[5488];
  __shared__ float fb[4];
  __shared__ unsigned char dep[22 * 22];
  __shared__ float li0[22 * 22], li1[22 * 22], li2[22 * 22];
  int t = threadIdx.x;
  for (int k = t; k < 5488; k += 256) fw[k] = f_w[k];
  if (t < 4) fb[t] = f_b[t];
  int bh = blockIdx.y * 16, bw = blockIdx.x * 16;
  for (int k = t; k < 484; k += 256) {
    int ty = k / 22, tx = k - ty * 22;
    int gy = bh - 3 + ty, gx = bw - 3 + tx;
    bool inb = (gy >= 0 && gy < HH && gx >= 0 && gx < WW);
    int gp = inb ? (gy * WW + gx) : 0;
    dep[k] = inb ? depth[gp] : (unsigned char)255;   // 255 -> never within |dz|<=3
    li0[k] = inb ? light[0 * HH * WW + gp] : 0.f;
    li1[k] = inb ? light[1 * HH * WW + gp] : 0.f;
    li2[k] = inb ? light[2 * HH * WW + gp] : 0.f;
  }
  __syncthreads();
  int ly = t >> 4, lx = t & 15;
  int h = bh + ly, w = bw + lx;
  int c = (ly + 3) * 22 + (lx + 3);
  int d0 = dep[c];
  float a0 = fb[0], a1 = fb[1], a2 = fb[2], a3 = fb[3];
#pragma unroll
  for (int dy = 0; dy < 7; ++dy) {
#pragma unroll
    for (int dx = 0; dx < 7; ++dx) {
      int n = c + (dy - 3) * 22 + (dx - 3);
      int dz = (int)dep[n] - d0 + 3;
      if ((unsigned)dz < 7u) {
        float wl0 = li0[n], wl1 = li1[n], wl2 = li2[n];
        int fo = dz * 49 + dy * 7 + dx;            // f_w[o][i] flat: (o*4+i)*343 + fo
        a0 += fw[0  * 343 + fo] * wl0 + fw[1  * 343 + fo] * wl1 + fw[2  * 343 + fo] * wl2 + fw[3  * 343 + fo];
        a1 += fw[4  * 343 + fo] * wl0 + fw[5  * 343 + fo] * wl1 + fw[6  * 343 + fo] * wl2 + fw[7  * 343 + fo];
        a2 += fw[8  * 343 + fo] * wl0 + fw[9  * 343 + fo] * wl1 + fw[10 * 343 + fo] * wl2 + fw[11 * 343 + fo];
        a3 += fw[12 * 343 + fo] * wl0 + fw[13 * 343 + fo] * wl1 + fw[14 * 343 + fo] * wl2 + fw[15 * 343 + fo];
      }
    }
  }
  int p = h * WW + w;
  float inv = 1.f / a3;
  out[0 * HH * WW + p] = a0 * inv * albedo[0 * HH * WW + p];
  out[1 * HH * WW + p] = a1 * inv * albedo[1 * HH * WW + p];
  out[2 * HH * WW + p] = a2 * inv * albedo[2 * HH * WW + p];
}

extern "C" void kernel_launch(void* const* d_in, const int* in_sizes, int n_in,
                              void* d_out, int out_size, void* d_ws, size_t ws_size,
                              hipStream_t stream) {
  const float* light   = (const float*)d_in[0];
  const float* albedo  = (const float*)d_in[1];
  const float* normals = (const float*)d_in[2];
  /* positions d_in[3] unused by the reference's dataflow */
  const float* c1_w = (const float*)d_in[4];
  const float* c1_b = (const float*)d_in[5];
  const float* g1 = (const float*)d_in[6],  *b1 = (const float*)d_in[7];
  const float* m1 = (const float*)d_in[8],  *v1 = (const float*)d_in[9];
  const float* c2_w = (const float*)d_in[10];
  const float* c2_b = (const float*)d_in[11];
  const float* g2 = (const float*)d_in[12], *b2 = (const float*)d_in[13];
  const float* m2 = (const float*)d_in[14], *v2 = (const float*)d_in[15];
  const float* c3_w = (const float*)d_in[16];
  const float* c3_b = (const float*)d_in[17];
  const float* g3 = (const float*)d_in[18], *b3 = (const float*)d_in[19];
  const float* m3 = (const float*)d_in[20], *v3 = (const float*)d_in[21];
  const float* f_w = (const float*)d_in[22];
  const float* f_b = (const float*)d_in[23];
  float* out = (float*)d_out;

  char* ws = (char*)d_ws;
  _Float16* xpc  = (_Float16*)(ws + O_XPC);
  _Float16* hp1c = (_Float16*)(ws + O_HP1);
  _Float16* hp2c = (_Float16*)(ws + O_HP2);
  _Float16* wb1  = (_Float16*)(ws + O_WB1);
  _Float16* wb2  = (_Float16*)(ws + O_WB2);
  float*    par  = (float*)   (ws + O_PAR);
  unsigned char* dep = (unsigned char*)(ws + O_DEP);

  dn_prep_params<<<1, 256, 0, stream>>>(c1_b, g1, b1, m1, v1,
                                        c2_b, g2, b2, m2, v2,
                                        c3_w, c3_b, g3, b3, m3, v3, par);
  dn_pack_xpc<<<((PP + PW) * 16 + 255) / 256, 256, 0, stream>>>(light, albedo, normals, xpc);
  {
    int n16 = (9634816 + 2408704) / 16;   // zero hp1c+hp2c (contiguous) -> SAME borders
    dn_zero16<<<(n16 + 255) / 256, 256, 0, stream>>>((uint4*)hp1c, n16);
  }
  dn_pack_wb1<<<(13312 + 255) / 256, 256, 0, stream>>>(c1_w, par, wb1);
  dn_pack_wb2<<<(12800 + 255) / 256, 256, 0, stream>>>(c2_w, par, wb2);

  dn_conv1_wmma<<<1152, 256, 0, stream>>>(xpc, wb1, par, hp1c);  // 9216 waves * 16 px
  dn_conv2_wmma<<<1152, 256, 0, stream>>>(hp1c, wb2, par, hp2c);
  dn_conv3_depth<<<576, 256, 0, stream>>>(hp2c, par, dep);
  dn_final<<<dim3(24, 24), 256, 0, stream>>>(dep, light, albedo, f_w, f_b, out);
  (void)in_sizes; (void)n_in; (void)out_size; (void)ws_size;
}
